// GAT_LSTMseq2seq_53412213293076
// MI455X (gfx1250) — compile-verified, matrix operand order checked by reading, not run on hardware
//
#include <hip/hip_runtime.h>
#include <hip/hip_bf16.h>

typedef __attribute__((ext_vector_type(16))) __bf16 v16bf;
typedef __attribute__((ext_vector_type(8)))  float  v8f;

static __device__ __forceinline__ float sigmf(float x) { return 1.0f / (1.0f + expf(-x)); }

// ---------------------------------------------------------------------------
// Pack f32 weights W[K,N] into bf16 transposed Wt[N,Kpad] (zero-padded K).
// ---------------------------------------------------------------------------
__global__ void wt_pack(const float* W, __bf16* Wt, int K, int Kpad, int Nw, int total) {
  int idx = blockIdx.x * 256 + threadIdx.x;
  if (idx >= total) return;
  int n = idx / Kpad;
  int k = idx - n * Kpad;
  Wt[idx] = (k < K) ? (__bf16)W[(size_t)k * Nw + n] : (__bf16)0.0f;
}

// ---------------------------------------------------------------------------
// Build bf16 input: [cont(ncont) | emb_T0[idx0]+emb_T1[idx1] | zero pad]
// ---------------------------------------------------------------------------
__global__ void build_input(const float* in, const float* T0, const float* T1,
                            __bf16* out, int rows, int in_w, int ncont, int emb_w,
                            int out_w, int total) {
  int idx = blockIdx.x * 256 + threadIdx.x;
  if (idx >= total) return;
  int row = idx / out_w;
  int col = idx - row * out_w;
  float v = 0.0f;
  if (col < ncont) {
    v = in[(size_t)row * in_w + col];
  } else if (col < ncont + emb_w) {
    int e  = col - ncont;
    int i0 = (int)in[(size_t)row * in_w + ncont];
    int i1 = (int)in[(size_t)row * in_w + ncont + 1];
    v = T0[i0 * emb_w + e] + T1[i1 * emb_w + e];
  }
  out[idx] = (__bf16)v;
}

// ---------------------------------------------------------------------------
// Generic WMMA GEMM: Out[M,Nout] = act(A[M,K] @ Wt^T + bias).
// Wt is [Nout,K] bf16 (pre-transposed). K mult of 32, M mult of 16,
// Nout mult of 128. Block = 256 thr = 8 waves, each wave one 16x16 tile.
// A tile (16 rows x K) staged once in dynamic LDS (16*K*2 bytes).
// In-place (A == Out, grid.y == 1) safe: all global A reads precede barrier.
// ---------------------------------------------------------------------------
template <int ACT>  // 0 = none, 1 = relu
__global__ void gemm_bf16(const __bf16* A, const __bf16* Wt, const float* bias,
                          __bf16* Out, int M, int K, int Nout) {
  extern __shared__ __bf16 Alds[];  // 16 * K
  const int tid = threadIdx.x;
  const int m0 = blockIdx.x * 16;
  const int cpr = K >> 4;  // 16-element chunks per row; 16*cpr = K <= 256
  if (tid < K) {
    int r = tid / cpr;
    int c = tid - r * cpr;
    *(v16bf*)&Alds[r * K + c * 16] =
        *(const v16bf*)(A + (size_t)(m0 + r) * K + c * 16);
  }
  __syncthreads();

  const int wave = tid >> 5;
  const int lane = tid & 31;
  const int lm = lane & 15, lh = lane >> 4;
  const int n0 = blockIdx.y * 128 + wave * 16;
  const size_t brow = (size_t)(n0 + lm) * K;
  v8f acc = {};
  for (int kt = 0; kt < K; kt += 32) {
    v16bf a = *(const v16bf*)&Alds[lm * K + kt + lh * 16];
    v16bf b = *(const v16bf*)(Wt + brow + kt + lh * 16);
    acc = __builtin_amdgcn_wmma_f32_16x16x32_bf16(false, a, false, b, (short)0, acc,
                                                  false, false);
  }
  const int col = n0 + lm;
  const float bv = bias ? bias[col] : 0.0f;
  for (int v = 0; v < 8; ++v) {
    int r = m0 + v + 8 * lh;
    float x = acc[v] + bv;
    if (ACT == 1) x = fmaxf(x, 0.0f);
    Out[(size_t)r * Nout + col] = (__bf16)x;
  }
}

// ---------------------------------------------------------------------------
// top/bot attention scores: per row dot(xe_row, a1/a2)
// ---------------------------------------------------------------------------
__global__ void attn_scores(const __bf16* xe, const float* a1, const float* a2,
                            float* top, float* bot, int rows) {
  int r = blockIdx.x * 256 + threadIdx.x;
  if (r >= rows) return;
  const __bf16* x = xe + (size_t)r * 128;
  float t = 0.0f, b = 0.0f;
  for (int c = 0; c < 128; ++c) {
    float v = (float)x[c];
    t += v * a1[c];
    b += v * a2[c];
  }
  top[r] = t;
  bot[r] = b;
}

// ---------------------------------------------------------------------------
// Transpose xe [bs][400][128] -> xeT [bs][128][416] (bf16, zero pad 400..415).
// Grid: (25 j-tiles, BS), block 256.
// ---------------------------------------------------------------------------
__global__ void xe_transpose(const __bf16* xe, __bf16* xeT) {
  __shared__ __bf16 t[16 * 128];
  const int tid = threadIdx.x;
  const int bs = blockIdx.y;
  const int j0 = blockIdx.x * 16;
  const size_t rbase = (size_t)bs * 400;
  for (int idx = tid; idx < 2048; idx += 256) {
    int jj = idx >> 7;   // 0..15
    int n  = idx & 127;
    t[n * 16 + jj] = xe[(rbase + j0 + jj) * 128 + n];
  }
  __syncthreads();
  for (int idx = tid; idx < 2048; idx += 256) {
    int n  = idx >> 4;
    int jl = idx & 15;
    xeT[((size_t)bs * 128 + n) * 416 + j0 + jl] = t[n * 16 + jl];
  }
  if (blockIdx.x == 24) {  // zero pad columns 400..415
    for (int idx = tid; idx < 2048; idx += 256) {
      int n  = idx >> 4;
      int jl = idx & 15;
      xeT[((size_t)bs * 128 + n) * 416 + 400 + jl] = (__bf16)0.0f;
    }
  }
}

// ---------------------------------------------------------------------------
// Fused GAT block: leaky(top_i+bot_j) -> adj mask -> softmax (16 rows in LDS)
// -> WMMA pi[16,416] @ xeT -> sigmoid -> Xseq (bf16).
// Grid: x = 25 i-tiles, y = B * S_part (one (b,s) per y).
// ---------------------------------------------------------------------------
__global__ void gat_fused(const __bf16* xeT, const float* top, const float* bot,
                          const float* adj, __bf16* xseq, int s_off, int S_part) {
  __shared__ float  pi_f[16 * 400];
  __shared__ float  red[16 * 16];
  __shared__ float  rowstat[16];
  __shared__ __bf16 pi_b[16 * 416];

  const int tid = threadIdx.x;
  const int bs = blockIdx.y;
  const int i0 = blockIdx.x * 16;
  const int bidx = bs / S_part;
  const int s_glob = s_off + (bs - bidx * S_part);

  const int il = tid >> 4;   // row within tile, 0..15
  const int sub = tid & 15;  // 16 threads cooperate per row
  const int i = i0 + il;
  const size_t rbase = (size_t)bs * 400;
  const float tv = top[rbase + i];

  // pass 1: z + mask + running max
  float lmax = -3.0e38f;
  for (int jj = sub; jj < 400; jj += 16) {
    float z = tv + bot[rbase + jj];
    z = (z > 0.0f) ? z : 0.01f * z;                // leaky relu
    if (adj[i * 400 + jj] == 0.0f) z = -3.0e38f;   // mask
    pi_f[il * 400 + jj] = z;
    lmax = fmaxf(lmax, z);
  }
  red[il * 16 + sub] = lmax;
  __syncthreads();
  if (sub == 0) {
    float m = red[il * 16];
    for (int t = 1; t < 16; ++t) m = fmaxf(m, red[il * 16 + t]);
    rowstat[il] = m;
  }
  __syncthreads();
  const float m = rowstat[il];
  // pass 2: exp + sum
  float lsum = 0.0f;
  for (int jj = sub; jj < 400; jj += 16) {
    float e = expf(pi_f[il * 400 + jj] - m);
    pi_f[il * 400 + jj] = e;
    lsum += e;
  }
  __syncthreads();  // protect rowstat before reuse
  red[il * 16 + sub] = lsum;
  __syncthreads();
  if (sub == 0) {
    float s = 0.0f;
    for (int t = 0; t < 16; ++t) s += red[il * 16 + t];
    rowstat[il] = s;
  }
  __syncthreads();
  const float inv = 1.0f / rowstat[il];
  // pass 3: normalized bf16 pi, zero pad 400..415
  for (int jj = sub; jj < 400; jj += 16)
    pi_b[il * 416 + jj] = (__bf16)(pi_f[il * 400 + jj] * inv);
  pi_b[il * 416 + 400 + sub] = (__bf16)0.0f;
  __syncthreads();

  // phase 2: WMMA pi[16,416] @ xeT[bs][128][416] -- branch-free K loop
  const int wave = tid >> 5;
  const int lane = tid & 31;
  const int lm = lane & 15, lh = lane >> 4;
  const int n0 = wave * 16;
  const size_t bbase = ((size_t)bs * 128 + (n0 + lm)) * 416;
  v8f acc = {};
  for (int kt = 0; kt < 416; kt += 32) {
    v16bf a = *(const v16bf*)&pi_b[lm * 416 + kt + lh * 16];
    v16bf b = *(const v16bf*)(xeT + bbase + kt + lh * 16);
    acc = __builtin_amdgcn_wmma_f32_16x16x32_bf16(false, a, false, b, (short)0, acc,
                                                  false, false);
  }
  const size_t obase = (size_t)s_glob * 6400 + (size_t)bidx * 400;
  for (int v = 0; v < 8; ++v) {
    int ii = i0 + v + 8 * lh;
    xseq[(obase + ii) * 128 + n0 + lm] = (__bf16)sigmf(acc[v]);
  }
}

// ---------------------------------------------------------------------------
// LSTM gate GEMM: [M,256] (xt||h) @ Wt_l[512,256]^T -> gates f32 [M,512]
// Grid: (M/16, 4), block 256. A tile staged in LDS.
// ---------------------------------------------------------------------------
__global__ void lstm_gates(const __bf16* xt, const __bf16* h, const __bf16* Wt,
                           float* gates, int M) {
  __shared__ __bf16 Alds[16 * 256];
  const int tid = threadIdx.x;
  const int m0 = blockIdx.x * 16;
  {
    int r = tid >> 4;        // 0..15
    int kk = (tid & 15) * 16;  // 0..240
    const __bf16* src = (kk < 128) ? (xt + (size_t)(m0 + r) * 128 + kk)
                                   : (h + (size_t)(m0 + r) * 128 + (kk - 128));
    *(v16bf*)&Alds[r * 256 + kk] = *(const v16bf*)src;
  }
  __syncthreads();

  const int wave = tid >> 5;
  const int lane = tid & 31;
  const int lm = lane & 15, lh = lane >> 4;
  const int n0 = blockIdx.y * 128 + wave * 16;
  const size_t brow = (size_t)(n0 + lm) * 256;
  v8f acc = {};
  for (int kt = 0; kt < 256; kt += 32) {
    v16bf a = *(const v16bf*)&Alds[lm * 256 + kt + lh * 16];
    v16bf b = *(const v16bf*)(Wt + brow + kt + lh * 16);
    acc = __builtin_amdgcn_wmma_f32_16x16x32_bf16(false, a, false, b, (short)0, acc,
                                                  false, false);
  }
  const int col = n0 + lm;
  for (int v = 0; v < 8; ++v) {
    int r = m0 + v + 8 * lh;
    gates[(size_t)r * 512 + col] = acc[v];
  }
}

__global__ void lstm_update(const float* gates, const float* bi, const float* bf,
                            const float* bo, const float* bg, float* c, __bf16* h,
                            __bf16* cb, int total) {
  int idx = blockIdx.x * 256 + threadIdx.x;
  if (idx >= total) return;
  int row = idx >> 7;
  int ch = idx & 127;
  const float* g = gates + (size_t)row * 512;
  float iv = sigmf(g[ch] + bi[ch]);
  float fv = sigmf(g[128 + ch] + bf[ch]);
  float ov = sigmf(g[256 + ch] + bo[ch]);
  float gv = tanhf(g[384 + ch] + bg[ch]);
  float cn = fv * c[idx] + iv * gv;
  c[idx] = cn;
  h[idx] = (__bf16)(ov * tanhf(cn));
  cb[idx] = (__bf16)cn;
}

// decoder final 128 -> 1, write out[b,n,t]
__global__ void dec_final(const __bf16* d2, const float* W3, const float* b3,
                          float* out, int t) {
  int row = blockIdx.x * 256 + threadIdx.x;
  if (row >= 6400) return;
  const __bf16* x = d2 + (size_t)row * 128;
  float s = b3[0];
  for (int c = 0; c < 128; ++c) s += (float)x[c] * W3[c];
  out[(size_t)row * 12 + t] = s;
}

// ---------------------------------------------------------------------------
extern "C" void kernel_launch(void* const* d_in, const int* in_sizes, int n_in,
                              void* d_out, int out_size, void* d_ws, size_t ws_size,
                              hipStream_t stream) {
  const float* x_past = (const float*)d_in[0];
  const float* x_fut  = (const float*)d_in[1];
  const float* adj    = (const float*)d_in[2];
  const float* emb_p0 = (const float*)d_in[3];
  const float* emb_p1 = (const float*)d_in[4];
  const float* emb_f0 = (const float*)d_in[5];
  const float* emb_f1 = (const float*)d_in[6];
  const float* pp_W1 = (const float*)d_in[7];
  const float* pp_b1 = (const float*)d_in[8];
  const float* pp_W2 = (const float*)d_in[9];
  const float* pp_b2 = (const float*)d_in[10];
  const float* pp_W3 = (const float*)d_in[11];
  const float* pf_W1 = (const float*)d_in[12];
  const float* pf_b1 = (const float*)d_in[13];
  const float* pf_W2 = (const float*)d_in[14];
  const float* pf_b2 = (const float*)d_in[15];
  const float* pf_W3 = (const float*)d_in[16];
  const float* gp_W = (const float*)d_in[17];
  const float* gp_a1 = (const float*)d_in[18];
  const float* gp_a2 = (const float*)d_in[19];
  const float* gf_W = (const float*)d_in[20];
  const float* gf_a1 = (const float*)d_in[21];
  const float* gf_a2 = (const float*)d_in[22];
  const float* Wi = (const float*)d_in[23];
  const float* bi = (const float*)d_in[24];
  const float* Wf = (const float*)d_in[25];
  const float* bf = (const float*)d_in[26];
  const float* Wo = (const float*)d_in[27];
  const float* bo = (const float*)d_in[28];
  const float* Wg = (const float*)d_in[29];
  const float* bg = (const float*)d_in[30];
  const float* dW1 = (const float*)d_in[31];
  const float* db1 = (const float*)d_in[32];
  const float* dW2 = (const float*)d_in[33];
  const float* db2 = (const float*)d_in[34];
  const float* dW3 = (const float*)d_in[35];
  const float* db3 = (const float*)d_in[36];
  float* out = (float*)d_out;

  constexpr int RP = 16 * 24 * 400;  // 153600 past rows
  constexpr int RF = 16 * 12 * 400;  // 76800 future rows
  constexpr int RS = 16 * 400;       // 6400 LSTM rows

  char* w = (char*)d_ws;
  size_t off = 0;
  auto alloc = [&](size_t bytes) -> char* {
    char* p = w + off;
    off += (bytes + 255) & ~(size_t)255;
    return p;
  };
  __bf16* Wt_pp1 = (__bf16*)alloc(128 * 96 * 2);
  __bf16* Wt_pp2 = (__bf16*)alloc(128 * 128 * 2);
  __bf16* Wt_pp3 = (__bf16*)alloc(128 * 128 * 2);
  __bf16* Wt_pf1 = (__bf16*)alloc(128 * 160 * 2);
  __bf16* Wt_pf2 = (__bf16*)alloc(128 * 128 * 2);
  __bf16* Wt_pf3 = (__bf16*)alloc(128 * 128 * 2);
  __bf16* Wt_gp = (__bf16*)alloc(128 * 128 * 2);
  __bf16* Wt_gf = (__bf16*)alloc(128 * 128 * 2);
  __bf16* Wt_l = (__bf16*)alloc(512 * 256 * 2);
  __bf16* Wt_d1 = (__bf16*)alloc(128 * 128 * 2);
  __bf16* Wt_d2 = (__bf16*)alloc(128 * 128 * 2);

  char* R1 = alloc((size_t)RP * 96 * 2);               // staging, reused by LSTM
  __bf16* H = (__bf16*)alloc((size_t)RP * 128 * 2);    // MLP ping (in-place)
  __bf16* XE = (__bf16*)alloc((size_t)RP * 128 * 2);   // GAT xe (row-major)
  __bf16* XET = (__bf16*)alloc((size_t)384 * 128 * 416 * 2);  // xe transposed+padded
  __bf16* Xseq = (__bf16*)alloc((size_t)36 * RS * 128 * 2);
  float* topP = (float*)alloc((size_t)RP * 4);
  float* botP = (float*)alloc((size_t)RP * 4);
  float* topF = (float*)alloc((size_t)RF * 4);
  float* botF = (float*)alloc((size_t)RF * 4);

  // LSTM-phase aliases inside R1 (free by then)
  float* gates = (float*)R1;                  // 6400*512*4 = 13107200
  float* cbuf = (float*)(R1 + 13107200);      // 6400*128*4
  __bf16* hbuf = (__bf16*)(R1 + 16384000);    // 6400*128*2
  __bf16* cb16 = (__bf16*)(R1 + 18022400);
  __bf16* dd1 = (__bf16*)(R1 + 19660800);
  __bf16* dd2 = (__bf16*)(R1 + 21299200);
  __bf16* Xp = (__bf16*)R1;
  __bf16* Xf = (__bf16*)R1;

  auto pack = [&](const float* W, __bf16* Wt, int K, int Kpad, int Nw) {
    int total = Nw * Kpad;
    wt_pack<<<(total + 255) / 256, 256, 0, stream>>>(W, Wt, K, Kpad, Nw, total);
  };
  pack(pp_W1, Wt_pp1, 70, 96, 128);
  pack(pp_W2, Wt_pp2, 128, 128, 128);
  pack(pp_W3, Wt_pp3, 128, 128, 128);
  pack(pf_W1, Wt_pf1, 130, 160, 128);
  pack(pf_W2, Wt_pf2, 128, 128, 128);
  pack(pf_W3, Wt_pf3, 128, 128, 128);
  pack(gp_W, Wt_gp, 128, 128, 128);
  pack(gf_W, Wt_gf, 128, 128, 128);
  pack(Wi, Wt_l + 0 * 128 * 256, 256, 256, 128);
  pack(Wf, Wt_l + 1 * 128 * 256, 256, 256, 128);
  pack(Wo, Wt_l + 2 * 128 * 256, 256, 256, 128);
  pack(Wg, Wt_l + 3 * 128 * 256, 256, 256, 128);
  pack(dW1, Wt_d1, 128, 128, 128);
  pack(dW2, Wt_d2, 128, 128, 128);

  // ---- past pipeline ----
  {
    int total = RP * 96;
    build_input<<<(total + 255) / 256, 256, 0, stream>>>(x_past, emb_p0, emb_p1, Xp,
                                                         RP, 8, 6, 64, 96, total);
    gemm_bf16<1><<<dim3(RP / 16, 1), 256, 16 * 96 * 2, stream>>>(Xp, Wt_pp1, pp_b1, H, RP, 96, 128);
    gemm_bf16<1><<<dim3(RP / 16, 1), 256, 16 * 128 * 2, stream>>>(H, Wt_pp2, pp_b2, H, RP, 128, 128);
    gemm_bf16<0><<<dim3(RP / 16, 1), 256, 16 * 128 * 2, stream>>>(H, Wt_pp3, nullptr, H, RP, 128, 128);
    gemm_bf16<0><<<dim3(RP / 16, 1), 256, 16 * 128 * 2, stream>>>(H, Wt_gp, nullptr, XE, RP, 128, 128);
    attn_scores<<<(RP + 255) / 256, 256, 0, stream>>>(XE, gp_a1, gp_a2, topP, botP, RP);
    xe_transpose<<<dim3(25, 16 * 24), 256, 0, stream>>>(XE, XET);
    gat_fused<<<dim3(25, 16 * 24), 256, 0, stream>>>(XET, topP, botP, adj, Xseq, 0, 24);
  }
  // ---- future pipeline (reuses R1/H/XE/XET) ----
  {
    int total = RF * 160;
    build_input<<<(total + 255) / 256, 256, 0, stream>>>(x_fut, emb_f0, emb_f1, Xf,
                                                         RF, 4, 2, 128, 160, total);
    gemm_bf16<1><<<dim3(RF / 16, 1), 256, 16 * 160 * 2, stream>>>(Xf, Wt_pf1, pf_b1, H, RF, 160, 128);
    gemm_bf16<1><<<dim3(RF / 16, 1), 256, 16 * 128 * 2, stream>>>(H, Wt_pf2, pf_b2, H, RF, 128, 128);
    gemm_bf16<0><<<dim3(RF / 16, 1), 256, 16 * 128 * 2, stream>>>(H, Wt_pf3, nullptr, H, RF, 128, 128);
    gemm_bf16<0><<<dim3(RF / 16, 1), 256, 16 * 128 * 2, stream>>>(H, Wt_gf, nullptr, XE, RF, 128, 128);
    attn_scores<<<(RF + 255) / 256, 256, 0, stream>>>(XE, gf_a1, gf_a2, topF, botF, RF);
    xe_transpose<<<dim3(25, 16 * 12), 256, 0, stream>>>(XE, XET);
    gat_fused<<<dim3(25, 16 * 12), 256, 0, stream>>>(XET, topF, botF, adj, Xseq, 24, 12);
  }
  // ---- LSTM + decoder (decoder only for s >= 24; earlier outputs discarded) ----
  hipMemsetAsync(cbuf, 0, (size_t)RS * 128 * 4, stream);
  hipMemsetAsync(hbuf, 0, (size_t)RS * 128 * 2, stream);
  for (int s = 0; s < 36; ++s) {
    const __bf16* xt = Xseq + (size_t)s * RS * 128;
    lstm_gates<<<dim3(RS / 16, 4), 256, 0, stream>>>(xt, hbuf, Wt_l, gates, RS);
    lstm_update<<<(RS * 128 + 255) / 256, 256, 0, stream>>>(gates, bi, bf, bo, bg,
                                                            cbuf, hbuf, cb16, RS * 128);
    if (s >= 24) {
      gemm_bf16<1><<<dim3(RS / 16, 1), 256, 16 * 128 * 2, stream>>>(cb16, Wt_d1, db1, dd1, RS, 128, 128);
      gemm_bf16<1><<<dim3(RS / 16, 1), 256, 16 * 128 * 2, stream>>>(dd1, Wt_d2, db2, dd2, RS, 128, 128);
      dec_final<<<(RS + 255) / 256, 256, 0, stream>>>(dd2, dW3, db3, out, s - 24);
    }
  }
  (void)in_sizes; (void)n_in; (void)out_size; (void)ws_size;
}
